// GCN_layer_18451179504149
// MI455X (gfx1250) — matrix operand order repaired and checked by reading, not compiled, not run on hardware
//
#include <hip/hip_runtime.h>

typedef __attribute__((ext_vector_type(16))) _Float16 v16h;
typedef __attribute__((ext_vector_type(8)))  float    v8f;

#define F_IN 128
#define HID  128
#define N_CLS 16

// ---------------------------------------------------------------------------
// Pack fp32 W[K][Nout] (row-major) into per-lane WMMA B-fragment layout (f16).
// Flat layout: Wpk[((nt*4 + ks)*32 + lane)*16 + e], where for the 16-bit B
// 32x16 tile: half = lane>>4, n = nt*16 + (lane&15), K = ks*32 + half*16 + e.
// ---------------------------------------------------------------------------
__global__ void pack_w_kernel(const float* __restrict__ W, _Float16* __restrict__ Wpk,
                              int nout_tiles, int nout) {
  int t = blockIdx.x * blockDim.x + threadIdx.x;
  int total = nout_tiles * 4 * 32 * 16;
  if (t >= total) return;
  int e      = t & 15;
  int lane   = (t >> 4) & 31;
  int ks     = (t >> 9) & 3;
  int nt     = t >> 11;
  int half16 = lane >> 4;
  int n = nt * 16 + (lane & 15);
  int k = ks * 32 + half16 * 16 + e;
  Wpk[t] = (_Float16)W[(size_t)k * nout + n];
}

// ---------------------------------------------------------------------------
// GEMM: C[nrows x NOUT] = A[nrows x 128] * W(packed).  One wave per 16-row
// strip; 4 waves per block.  f16 WMMA with f32 accumulate.
// ---------------------------------------------------------------------------
template <int NOUT>
__global__ __launch_bounds__(128)
void gemm_wmma_kernel(const float* __restrict__ A, const _Float16* __restrict__ Wpk,
                      float* __restrict__ C, int nrows) {
  constexpr int LDST = 132;                 // padded row stride (floats)
  __shared__ float As[4 * 16 * LDST];

  const int lane  = threadIdx.x & 31;
  const int wv    = threadIdx.x >> 5;
  const int strip = blockIdx.x * 4 + wv;
  const int m0    = strip * 16;
  if (m0 >= nrows) return;                  // uniform per wave

  float* myLds = &As[wv * 16 * LDST];
  {
    // Coalesced stage of the 16x128 fp32 strip into LDS.
    const float4* src = reinterpret_cast<const float4*>(A + (size_t)m0 * 128);
#pragma unroll
    for (int i = 0; i < 16; ++i) {
      int f   = i * 32 + lane;              // 512 float4 per strip
      int row = f >> 5;
      int c4  = f & 31;
      float4 val = src[row * 32 + c4];
      *reinterpret_cast<float4*>(&myLds[row * LDST + c4 * 4]) = val;
    }
  }
  __syncthreads();

  const int half16 = lane >> 4;
  const int m      = lane & 15;

  // Pack A fragments (contiguous k = half*16 + e, matching packed B), 4 k-steps of K=32.
  v16h afrag[4];
#pragma unroll
  for (int ks = 0; ks < 4; ++ks) {
#pragma unroll
    for (int v = 0; v < 8; ++v) {
      int kb = ks * 32 + 16 * half16 + 2 * v;
      float2 x = *reinterpret_cast<const float2*>(&myLds[m * LDST + kb]);
      afrag[ks][2 * v]     = (_Float16)x.x;
      afrag[ks][2 * v + 1] = (_Float16)x.y;
    }
  }

#pragma unroll
  for (int nt = 0; nt < NOUT / 16; ++nt) {
    v8f c = {};
#pragma unroll
    for (int ks = 0; ks < 4; ++ks) {
      v16h b = *reinterpret_cast<const v16h*>(Wpk + ((size_t)((nt * 4 + ks) * 32 + lane)) * 16);
      c = __builtin_amdgcn_wmma_f32_16x16x32_f16(false, afrag[ks], false, b,
                                                 (short)0, c, false, false);
    }
#pragma unroll
    for (int r = 0; r < 8; ++r) {
      C[(size_t)(m0 + r + 8 * half16) * NOUT + nt * 16 + m] = c[r];
    }
  }
}

// ---------------------------------------------------------------------------
// Degree / normalization
// ---------------------------------------------------------------------------
__global__ void deg_init_kernel(float* __restrict__ deg, int n) {
  int i = blockIdx.x * blockDim.x + threadIdx.x;
  if (i < n) deg[i] = 1.0f;                 // self loop
}

__global__ void deg_edges_kernel(const int* __restrict__ col, float* __restrict__ deg,
                                 int nedges) {
  int t = blockIdx.x * blockDim.x + threadIdx.x;
  if (t < nedges) atomicAdd(&deg[col[t]], 1.0f);
}

__global__ void dinv_kernel(float* __restrict__ deg, int n) {
  int i = blockIdx.x * blockDim.x + threadIdx.x;
  if (i < n) deg[i] = rsqrtf(deg[i]);       // deg >= 1 always
}

// ---------------------------------------------------------------------------
// agg[i] = dinv[i]^2 * h[i]   (self-loop term; also initializes agg)
// ---------------------------------------------------------------------------
template <int D>
__global__ void init_agg_kernel(const float* __restrict__ h, const float* __restrict__ dinv,
                                float* __restrict__ agg, int n) {
  int t = blockIdx.x * blockDim.x + threadIdx.x;
  int total = n * (D / 4);
  if (t >= total) return;
  int node = t / (D / 4);
  float s = dinv[node];
  s = s * s;
  float4 v = reinterpret_cast<const float4*>(h)[t];
  float4 o = {v.x * s, v.y * s, v.z * s, v.w * s};
  reinterpret_cast<float4*>(agg)[t] = o;
}

// ---------------------------------------------------------------------------
// Edge scatter: agg[col] += dinv[row]*dinv[col] * h[row].  D/4 lanes per edge.
// ---------------------------------------------------------------------------
template <int D>
__global__ void scatter_kernel(const float* __restrict__ h, const int* __restrict__ row,
                               const int* __restrict__ col, const float* __restrict__ dinv,
                               float* __restrict__ agg, int nedges) {
  constexpr int LPE = D / 4;
  long long t = (long long)blockIdx.x * blockDim.x + threadIdx.x;
  int e = (int)(t / LPE);
  int l = (int)(t % LPE);
  if (e >= nedges) return;
  int r = row[e];
  int c = col[e];
  float nrm = dinv[r] * dinv[c];
  float4 v = reinterpret_cast<const float4*>(h + (size_t)r * D)[l];
  float* dst = agg + (size_t)c * D + l * 4;
  atomicAdd(dst + 0, v.x * nrm);
  atomicAdd(dst + 1, v.y * nrm);
  atomicAdd(dst + 2, v.z * nrm);
  atomicAdd(dst + 3, v.w * nrm);
}

// ---------------------------------------------------------------------------
// x = relu(x + b)  in place
// ---------------------------------------------------------------------------
template <int D>
__global__ void bias_relu_kernel(float* __restrict__ x, const float* __restrict__ b, int n) {
  int t = blockIdx.x * blockDim.x + threadIdx.x;
  int total = n * D;
  if (t >= total) return;
  float v = x[t] + b[t & (D - 1)];
  x[t] = v > 0.0f ? v : 0.0f;
}

// ---------------------------------------------------------------------------
static inline int cdiv(long long a, int b) { return (int)((a + b - 1) / b); }

extern "C" void kernel_launch(void* const* d_in, const int* in_sizes, int n_in,
                              void* d_out, int out_size, void* d_ws, size_t ws_size,
                              hipStream_t stream) {
  const float* x  = (const float*)d_in[0];
  const int*   ei = (const int*)d_in[1];
  const float* W1 = (const float*)d_in[2];
  const float* b1 = (const float*)d_in[3];
  const float* W2 = (const float*)d_in[4];
  const float* b2 = (const float*)d_in[5];
  const float* W3 = (const float*)d_in[6];
  const float* b3 = (const float*)d_in[7];
  float* out = (float*)d_out;

  const int n      = in_sizes[0] / F_IN;    // 50000
  const int nedges = in_sizes[1] / 2;       // 1600000
  const int* erow = ei;                     // sources (row)
  const int* ecol = ei + nedges;            // targets (col)

  // workspace carve-up
  char* ws = (char*)d_ws;
  size_t off = 0;
  auto carve = [&](size_t bytes) -> void* {
    void* p = ws + off;
    off += (bytes + 255) & ~(size_t)255;
    return p;
  };
  _Float16* W1p  = (_Float16*)carve((size_t)8 * 4 * 32 * 16 * sizeof(_Float16));
  _Float16* W2p  = (_Float16*)carve((size_t)8 * 4 * 32 * 16 * sizeof(_Float16));
  _Float16* W3p  = (_Float16*)carve((size_t)1 * 4 * 32 * 16 * sizeof(_Float16));
  float*    dinv = (float*)carve((size_t)n * sizeof(float));
  float*    bufA = (float*)carve((size_t)n * HID * sizeof(float));
  float*    bufB = (float*)carve((size_t)n * HID * sizeof(float));
  float*    bufC = (float*)carve((size_t)n * N_CLS * sizeof(float));

  const int B = 256;

  // ---- weight packing (once per launch; tiny) ----
  pack_w_kernel<<<cdiv(8 * 2048, B), B, 0, stream>>>(W1, W1p, 8, HID);
  pack_w_kernel<<<cdiv(8 * 2048, B), B, 0, stream>>>(W2, W2p, 8, HID);
  pack_w_kernel<<<cdiv(1 * 2048, B), B, 0, stream>>>(W3, W3p, 1, N_CLS);

  // ---- gcn_norm: deg -> dinv ----
  deg_init_kernel<<<cdiv(n, B), B, 0, stream>>>(dinv, n);
  deg_edges_kernel<<<cdiv(nedges, B), B, 0, stream>>>(ecol, dinv, nedges);
  dinv_kernel<<<cdiv(n, B), B, 0, stream>>>(dinv, n);

  const int strips     = cdiv(n, 16);
  const int gemmBlocks = cdiv(strips, 4);

  // ---- layer 1 ----
  gemm_wmma_kernel<HID><<<gemmBlocks, 128, 0, stream>>>(x, W1p, bufA, n);
  init_agg_kernel<HID><<<cdiv((long long)n * 32, B), B, 0, stream>>>(bufA, dinv, bufB, n);
  scatter_kernel<HID><<<cdiv((long long)nedges * 32, B), B, 0, stream>>>(bufA, erow, ecol, dinv, bufB, nedges);
  bias_relu_kernel<HID><<<cdiv((long long)n * HID, B), B, 0, stream>>>(bufB, b1, n);

  // ---- layer 2 ----
  gemm_wmma_kernel<HID><<<gemmBlocks, 128, 0, stream>>>(bufB, W2p, bufA, n);
  init_agg_kernel<HID><<<cdiv((long long)n * 32, B), B, 0, stream>>>(bufA, dinv, bufB, n);
  scatter_kernel<HID><<<cdiv((long long)nedges * 32, B), B, 0, stream>>>(bufA, erow, ecol, dinv, bufB, nedges);
  bias_relu_kernel<HID><<<cdiv((long long)n * HID, B), B, 0, stream>>>(bufB, b2, n);

  // ---- layer 3 (output dim 16) ----
  gemm_wmma_kernel<N_CLS><<<gemmBlocks, 128, 0, stream>>>(bufB, W3p, bufC, n);
  init_agg_kernel<N_CLS><<<cdiv((long long)n * 4, B), B, 0, stream>>>(bufC, dinv, out, n);
  scatter_kernel<N_CLS><<<cdiv((long long)nedges * 4, B), B, 0, stream>>>(bufC, erow, ecol, dinv, out, nedges);
  bias_relu_kernel<N_CLS><<<cdiv((long long)n * N_CLS, B), B, 0, stream>>>(out, b3, n);
}
